// DecoderRNN_72249939854008
// MI455X (gfx1250) — compile-verified, matrix operand order checked by reading, not run on hardware
//
#include <hip/hip_runtime.h>
#include <hip/hip_bf16.h>

// Problem constants (from reference)
#define B_  64
#define P_  196
#define E_  2048
#define M_  512
#define D_  1024
#define A_  512
#define V_  10000
#define T_  22
#define TM1 21
#define KC  (M_ + E_ + D_)   // 3584: concat K for fused LSTM gate GEMM

typedef __attribute__((ext_vector_type(16))) __bf16 v16bf;
typedef __attribute__((ext_vector_type(8)))  float  v8f;

__device__ __forceinline__ float sigmoidf_(float x) { return 1.f / (1.f + __expf(-x)); }

// ---------------------------------------------------------------------------
// WMMA fragment loader.
// CDNA5 16-bit A-fragment layout (wave32): lane L holds row M=L&15; elems 0..7
// are K = kb+0..7 and elems 8..15 are K = 16+kb+0..7 with kb = (L>=16)?8:0.
// B is supplied as B^T (N x K row-major), which by M<->N symmetry uses the
// exact same per-lane pattern -> two contiguous 16-byte loads per fragment.
// ---------------------------------------------------------------------------
__device__ __forceinline__ v16bf frag_load(const __bf16* tile, int ld, int lane) {
  const int row = lane & 15;
  const int kb  = (lane & 16) >> 1;          // 0 or 8
  const __bf16* p = tile + (size_t)row * ld + kb;
  union { v16bf v; uint4 q[2]; } u;
  u.q[0] = *reinterpret_cast<const uint4*>(p);
  u.q[1] = *reinterpret_cast<const uint4*>(p + 16);
  return u.v;
}

// ---------------------------------------------------------------------------
// Generic bf16 WMMA GEMM:  D[M,N] = A[M,K] * BT[N,K]^T (+bias) (+sigmoid)
// (+row mask from declen/t -> zeros) (+optional bf16 mirror of D).
// Block = 256 threads = 8 waves; wave (wm=wid&3, wn=wid>>2) computes a 16x64
// strip via 4 accumulators sharing one A fragment. Block tile = 64 x 128.
// Requires M % 64 == 0 (true for all call sites: 64 and 12544), K % 32 == 0,
// N % 16 == 0. Ragged N at 128-tile granularity is handled by CLAMPING the
// OOB 16-wide n-tiles onto the last valid tile (redundant compute) so that
// no branch ever wraps a WMMA: EXEC stays all-ones as the ISA requires.
// All five fragments of a K-chunk are loaded into distinct values BEFORE any
// WMMA is issued, so the backend batches the loads into one clause and the
// WMMAs run back-to-back instead of load->waitcnt0->wmma serialization.
// ---------------------------------------------------------------------------
__global__ __launch_bounds__(256)
void gemm_bf16_wmma(const __bf16* __restrict__ A, int lda,
                    const __bf16* __restrict__ BT, int ldb,
                    float* __restrict__ D, long long ldd,
                    const float* __restrict__ bias,
                    const int* __restrict__ declen, int tstep,
                    int M, int N, int K, int act,
                    __bf16* __restrict__ Dbf, long long lddbf)
{
  const int lane = threadIdx.x & 31;
  const int wid  = threadIdx.x >> 5;
  const int m0 = blockIdx.y * 64 + (wid & 3) * 16;
  const int n0 = blockIdx.x * 128 + (wid >> 2) * 64;

  const __bf16* Arow = A + (size_t)m0 * lda;
  const __bf16* B0;
  const __bf16* B1;
  const __bf16* B2;
  const __bf16* B3;
  {
    int t0 = n0;      if (t0 + 16 > N) t0 = N - 16;   // clamp OOB tiles: redundant
    int t1 = n0 + 16; if (t1 + 16 > N) t1 = N - 16;   // compute, never a branch
    int t2 = n0 + 32; if (t2 + 16 > N) t2 = N - 16;   // around a WMMA
    int t3 = n0 + 48; if (t3 + 16 > N) t3 = N - 16;
    B0 = BT + (size_t)t0 * ldb;
    B1 = BT + (size_t)t1 * ldb;
    B2 = BT + (size_t)t2 * ldb;
    B3 = BT + (size_t)t3 * ldb;
  }

  v8f acc0 = (v8f){0.f,0.f,0.f,0.f,0.f,0.f,0.f,0.f};
  v8f acc1 = acc0, acc2 = acc0, acc3 = acc0;

  for (int k0 = 0; k0 < K; k0 += 32) {
    // Load all fragments first (distinct values -> distinct VGPR quads -> one
    // load clause + overlapped latency), then issue the 4 WMMAs together.
    v16bf a  = frag_load(Arow + k0, lda, lane);
    v16bf b0 = frag_load(B0   + k0, ldb, lane);
    v16bf b1 = frag_load(B1   + k0, ldb, lane);
    v16bf b2 = frag_load(B2   + k0, ldb, lane);
    v16bf b3 = frag_load(B3   + k0, ldb, lane);
    if (k0 + 64 <= K)  // stream-ahead prefetch of next A chunk (global_prefetch_b8)
      __builtin_prefetch(Arow + (size_t)(lane & 15) * lda + k0 + 32, 0, 1);
    acc0 = __builtin_amdgcn_wmma_f32_16x16x32_bf16(false, a, false, b0, (short)0, acc0, false, false);
    acc1 = __builtin_amdgcn_wmma_f32_16x16x32_bf16(false, a, false, b1, (short)0, acc1, false, false);
    acc2 = __builtin_amdgcn_wmma_f32_16x16x32_bf16(false, a, false, b2, (short)0, acc2, false, false);
    acc3 = __builtin_amdgcn_wmma_f32_16x16x32_bf16(false, a, false, b3, (short)0, acc3, false, false);
  }

  // C/D layout: VGPR r, lane L -> M = m0 + r + (L>=16?8:0), N = n0+i*16 + (L&15)
  const int mo = (lane & 16) >> 1;   // 0 or 8
  const int nc = lane & 15;
  v8f accs[4] = {acc0, acc1, acc2, acc3};
  #pragma unroll
  for (int i = 0; i < 4; ++i) {
    const int n = n0 + i * 16 + nc;
    if (n < N) {                      // store predication only (no WMMA inside)
      const float bv = bias ? bias[n] : 0.f;
      #pragma unroll
      for (int r = 0; r < 8; ++r) {
        const int m = m0 + mo + r;
        float v = accs[i][r] + bv;
        if (act == 1) v = sigmoidf_(v);
        if (declen && !(tstep < declen[m])) v = 0.f;
        D[(size_t)m * ldd + n] = v;
        if (Dbf) Dbf[(size_t)m * lddbf + n] = (__bf16)v;
      }
    }
  }
}

// ---------------------------------------------------------------------------
// Stable argsort of 64 lengths by descending length (ties: lower index first),
// matching jnp.argsort(-lengths). Also emits dec_len and float outputs.
// ---------------------------------------------------------------------------
__global__ void sort_kernel(const int* __restrict__ cap_len,
                            int* __restrict__ sortind, int* __restrict__ declen,
                            float* __restrict__ out_sortind, float* __restrict__ out_declen)
{
  const int i = threadIdx.x;            // 64 threads
  __shared__ int L[B_];
  const int li = cap_len[i];
  L[i] = li;
  __syncthreads();
  int rank = 0;
  for (int j = 0; j < B_; ++j) {
    const int lj = L[j];
    if (lj > li || (lj == li && j < i)) ++rank;
  }
  sortind[rank]     = i;
  declen[rank]      = li - 1;
  out_sortind[rank] = (float)i;
  out_declen[rank]  = (float)(li - 1);
}

__global__ void caps_kernel(const int* __restrict__ captions, const int* __restrict__ sortind,
                            int* __restrict__ caps_s, float* __restrict__ out_caps)
{
  const int idx = blockIdx.x * blockDim.x + threadIdx.x;
  if (idx >= B_ * T_) return;
  const int r = idx / T_, k = idx % T_;
  const int v = captions[sortind[r] * T_ + k];
  caps_s[idx]   = v;
  out_caps[idx] = (float)v;
}

__global__ void embs_kernel(const float* __restrict__ emb, const int* __restrict__ caps_s,
                            __bf16* __restrict__ out)
{
  const int idx = blockIdx.x * blockDim.x + threadIdx.x;
  if (idx >= B_ * T_ * M_) return;
  const int rt = idx / M_, m = idx % M_;
  const int tok = caps_s[rt];
  out[idx] = (__bf16)emb[(size_t)tok * M_ + m];
}

__global__ void enc_gather_kernel(const float* __restrict__ enc, const int* __restrict__ sortind,
                                  __bf16* __restrict__ out)
{
  const size_t total  = (size_t)B_ * P_ * E_;
  const size_t stride = (size_t)gridDim.x * blockDim.x;
  for (size_t i = (size_t)blockIdx.x * blockDim.x + threadIdx.x; i < total; i += stride) {
    const int r = (int)(i / ((size_t)P_ * E_));
    const size_t rem = i - (size_t)r * P_ * E_;
    out[i] = (__bf16)enc[(size_t)sortind[r] * P_ * E_ + rem];
  }
}

__global__ void mean_kernel(const float* __restrict__ enc, const int* __restrict__ sortind,
                            __bf16* __restrict__ mean_bf)
{
  const int r = blockIdx.x;
  const float* e = enc + (size_t)sortind[r] * P_ * E_;
  for (int c = threadIdx.x; c < E_; c += blockDim.x) {
    float s = 0.f;
    for (int p = 0; p < P_; ++p) s += e[(size_t)p * E_ + c];
    mean_bf[(size_t)r * E_ + c] = (__bf16)(s * (1.f / (float)P_));
  }
}

// in (K,N) f32 row-major -> out (N,K) bf16 row-major (weight -> B^T layout)
__global__ void transpose_bf16_kernel(const float* __restrict__ in, __bf16* __restrict__ out,
                                      int K, int N)
{
  const size_t total  = (size_t)K * N;
  const size_t stride = (size_t)gridDim.x * blockDim.x;
  for (size_t i = (size_t)blockIdx.x * blockDim.x + threadIdx.x; i < total; i += stride) {
    const int k = (int)(i / N), n = (int)(i % N);
    out[(size_t)n * K + k] = (__bf16)in[i];
  }
}

// Wcomb[n][0:2560]=W_ih[n], Wcomb[n][2560:3584]=W_hh[n]; bcomb = b_ih + b_hh
__global__ void build_wcomb_kernel(const float* __restrict__ Wih, const float* __restrict__ Whh,
                                   const float* __restrict__ bih, const float* __restrict__ bhh,
                                   __bf16* __restrict__ wcomb, float* __restrict__ bcomb)
{
  const size_t total  = (size_t)4 * D_ * KC;
  const size_t stride = (size_t)gridDim.x * blockDim.x;
  for (size_t i = (size_t)blockIdx.x * blockDim.x + threadIdx.x; i < total; i += stride) {
    const int n = (int)(i / KC), k = (int)(i % KC);
    const float v = (k < (M_ + E_)) ? Wih[(size_t)n * (M_ + E_) + k]
                                    : Whh[(size_t)n * D_ + (k - (M_ + E_))];
    wcomb[i] = (__bf16)v;
    if (i < (size_t)4 * D_) bcomb[i] = bih[i] + bhh[i];
  }
}

// ---------------------------------------------------------------------------
// Attention step (one block per batch row r):
//   e[p] = sum_a relu(att1[r,p,a] + att2[r,a]) * w_full[a] + b_full
//   alpha = softmax_p(e);  awe[r,:] = sum_p alpha[p] * enc[src,p,:]
// Writes masked alphas to d_out.
// ---------------------------------------------------------------------------
__global__ __launch_bounds__(256)
void attention_step(const float* __restrict__ att1, const float* __restrict__ att2,
                    const float* __restrict__ wfull, const float* __restrict__ bfull_p,
                    const float* __restrict__ enc, const int* __restrict__ sortind,
                    const int* __restrict__ declen, int t,
                    float* __restrict__ awe, float* __restrict__ alphas_out)
{
  const int r = blockIdx.x;
  __shared__ float att2s[A_], wfs[A_], ev[256], red[256];
  for (int i = threadIdx.x; i < A_; i += 256) { att2s[i] = att2[(size_t)r * A_ + i]; wfs[i] = wfull[i]; }
  __syncthreads();

  const int wid = threadIdx.x >> 5, lane = threadIdx.x & 31;
  const float bfull = bfull_p[0];

  for (int p = wid; p < P_; p += 8) {
    const float* row = att1 + ((size_t)r * P_ + p) * A_;
    float s = 0.f;
    for (int a = lane; a < A_; a += 32) {
      const float x = row[a] + att2s[a];
      s += fmaxf(x, 0.f) * wfs[a];
    }
    for (int o = 16; o; o >>= 1) s += __shfl_down(s, o, 32);   // wave32 reduce
    if (lane == 0) ev[p] = s + bfull;
  }
  __syncthreads();

  // softmax over P_=196
  float mx = -1e30f;
  for (int p = threadIdx.x; p < P_; p += 256) mx = fmaxf(mx, ev[p]);
  red[threadIdx.x] = mx; __syncthreads();
  for (int s2 = 128; s2; s2 >>= 1) {
    if (threadIdx.x < s2) red[threadIdx.x] = fmaxf(red[threadIdx.x], red[threadIdx.x + s2]);
    __syncthreads();
  }
  mx = red[0]; __syncthreads();

  float sm = 0.f;
  for (int p = threadIdx.x; p < P_; p += 256) { const float e = __expf(ev[p] - mx); ev[p] = e; sm += e; }
  red[threadIdx.x] = sm; __syncthreads();
  for (int s2 = 128; s2; s2 >>= 1) {
    if (threadIdx.x < s2) red[threadIdx.x] += red[threadIdx.x + s2];
    __syncthreads();
  }
  const float inv = 1.f / red[0]; __syncthreads();

  const bool on = t < declen[r];
  for (int p = threadIdx.x; p < P_; p += 256) {
    const float al = ev[p] * inv;
    ev[p] = al;
    alphas_out[((size_t)r * TM1 + t) * P_ + p] = on ? al : 0.f;
  }
  __syncthreads();

  // awe: fp32 reduction straight from the (L2-resident) original encoder data
  const float* encb = enc + (size_t)sortind[r] * P_ * E_;
  for (int e0 = threadIdx.x; e0 < E_; e0 += 256) {
    float s = 0.f;
    for (int p = 0; p < P_; ++p) s += ev[p] * encb[(size_t)p * E_ + e0];
    awe[(size_t)r * E_ + e0] = s;
  }
}

// xh row: [ e_t (512) | gate*awe (2048) | h (1024) ]  in bf16
__global__ void build_x_kernel(const __bf16* __restrict__ embs, const float* __restrict__ gate,
                               const float* __restrict__ awe, const __bf16* __restrict__ hbf,
                               __bf16* __restrict__ xh, int t)
{
  const int idx = blockIdx.x * blockDim.x + threadIdx.x;
  if (idx >= B_ * KC) return;
  const int r = idx / KC, k = idx % KC;
  __bf16 v;
  if (k < M_)            v = embs[((size_t)r * T_ + t) * M_ + k];
  else if (k < M_ + E_)  { const int e = k - M_; v = (__bf16)(gate[(size_t)r * E_ + e] * awe[(size_t)r * E_ + e]); }
  else                   v = hbf[(size_t)r * D_ + (k - M_ - E_)];
  xh[idx] = v;
}

__global__ void lstm_step(const float* __restrict__ gates, float* __restrict__ h,
                          float* __restrict__ c, __bf16* __restrict__ hbf,
                          const int* __restrict__ declen, int t)
{
  const int idx = blockIdx.x * blockDim.x + threadIdx.x;
  if (idx >= B_ * D_) return;
  const int r = idx >> 10, d = idx & (D_ - 1);
  const float* g = gates + (size_t)r * 4 * D_;
  const float ig = sigmoidf_(g[d]);
  const float fg = sigmoidf_(g[D_ + d]);
  const float gg = tanhf(g[2 * D_ + d]);
  const float og = sigmoidf_(g[3 * D_ + d]);
  const float cn = fg * c[idx] + ig * gg;
  const float hn = og * tanhf(cn);
  const bool on = t < declen[r];
  const float h2 = on ? hn : h[idx];
  const float c2 = on ? cn : c[idx];
  h[idx] = h2; c[idx] = c2; hbf[idx] = (__bf16)h2;
}

// ---------------------------------------------------------------------------
// Host orchestration
// ---------------------------------------------------------------------------
static inline void launch_gemm(hipStream_t s,
                               const __bf16* A, int lda, const __bf16* BT, int ldb,
                               float* D, long long ldd, const float* bias,
                               const int* declen, int t,
                               int M, int N, int K, int act,
                               __bf16* Dbf, long long lddbf)
{
  dim3 grid((N + 127) / 128, (M + 63) / 64);
  gemm_bf16_wmma<<<grid, 256, 0, s>>>(A, lda, BT, ldb, D, ldd, bias, declen, t,
                                      M, N, K, act, Dbf, lddbf);
}

extern "C" void kernel_launch(void* const* d_in, const int* in_sizes, int n_in,
                              void* d_out, int out_size, void* d_ws, size_t ws_size,
                              hipStream_t stream)
{
  const float* encoder_out = (const float*)d_in[0];
  const int*   captions    = (const int*)  d_in[1];
  const int*   cap_len     = (const int*)  d_in[2];
  const float* emb         = (const float*)d_in[3];
  const float* W_enc_att   = (const float*)d_in[4];
  const float* b_enc_att   = (const float*)d_in[5];
  const float* W_dec_att   = (const float*)d_in[6];
  const float* b_dec_att   = (const float*)d_in[7];
  const float* w_full      = (const float*)d_in[8];
  const float* b_full      = (const float*)d_in[9];
  const float* W_ih        = (const float*)d_in[10];
  const float* b_ih        = (const float*)d_in[11];
  const float* W_hh        = (const float*)d_in[12];
  const float* b_hh        = (const float*)d_in[13];
  const float* W_init_h    = (const float*)d_in[14];
  const float* b_init_h    = (const float*)d_in[15];
  const float* W_init_c    = (const float*)d_in[16];
  const float* b_init_c    = (const float*)d_in[17];
  const float* W_fbeta     = (const float*)d_in[18];
  const float* b_fbeta     = (const float*)d_in[19];
  const float* W_fc        = (const float*)d_in[20];
  const float* b_fc        = (const float*)d_in[21];
  (void)in_sizes; (void)n_in; (void)out_size; (void)ws_size;

  float* out         = (float*)d_out;
  float* out_pred    = out;                                   // (B, T-1, V)
  float* out_caps    = out_pred + (size_t)B_ * TM1 * V_;      // (B, T)
  float* out_declen  = out_caps + (size_t)B_ * T_;            // (B)
  float* out_alphas  = out_declen + B_;                       // (B, T-1, P)
  float* out_sortind = out_alphas + (size_t)B_ * TM1 * P_;    // (B)

  // workspace carving (256B aligned)
  char* wbase = (char*)d_ws;
  size_t off = 0;
  auto alloc = [&](size_t bytes) -> void* {
    void* p = wbase + off;
    off = (off + bytes + 255) & ~(size_t)255;
    return p;
  };
  int*    sortind  = (int*)   alloc((size_t)B_ * 4);
  int*    declen   = (int*)   alloc((size_t)B_ * 4);
  int*    caps_s   = (int*)   alloc((size_t)B_ * T_ * 4);
  __bf16* embs_bf  = (__bf16*)alloc((size_t)B_ * T_ * M_ * 2);
  __bf16* enc_bf   = (__bf16*)alloc((size_t)B_ * P_ * E_ * 2);
  __bf16* mean_bf  = (__bf16*)alloc((size_t)B_ * E_ * 2);
  __bf16* Wenc_t   = (__bf16*)alloc((size_t)A_ * E_ * 2);
  __bf16* Wdec_t   = (__bf16*)alloc((size_t)A_ * D_ * 2);
  __bf16* Wfbeta_t = (__bf16*)alloc((size_t)E_ * D_ * 2);
  __bf16* Wfc_t    = (__bf16*)alloc((size_t)V_ * D_ * 2);
  __bf16* Winith_t = (__bf16*)alloc((size_t)D_ * E_ * 2);
  __bf16* Winitc_t = (__bf16*)alloc((size_t)D_ * E_ * 2);
  __bf16* Wcomb    = (__bf16*)alloc((size_t)4 * D_ * KC * 2);
  float*  bcomb    = (float*) alloc((size_t)4 * D_ * 4);
  float*  h_f      = (float*) alloc((size_t)B_ * D_ * 4);
  float*  c_f      = (float*) alloc((size_t)B_ * D_ * 4);
  __bf16* h_bf     = (__bf16*)alloc((size_t)B_ * D_ * 2);
  float*  att1     = (float*) alloc((size_t)B_ * P_ * A_ * 4);
  float*  att2     = (float*) alloc((size_t)B_ * A_ * 4);
  float*  awe      = (float*) alloc((size_t)B_ * E_ * 4);
  float*  gateb    = (float*) alloc((size_t)B_ * E_ * 4);
  __bf16* xh       = (__bf16*)alloc((size_t)B_ * KC * 2);
  float*  gatesb   = (float*) alloc((size_t)B_ * 4 * D_ * 4);

  // ---- prep ----
  sort_kernel<<<1, B_, 0, stream>>>(cap_len, sortind, declen, out_sortind, out_declen);
  caps_kernel<<<(B_ * T_ + 255) / 256, 256, 0, stream>>>(captions, sortind, caps_s, out_caps);
  embs_kernel<<<(B_ * T_ * M_ + 255) / 256, 256, 0, stream>>>(emb, caps_s, embs_bf);
  enc_gather_kernel<<<4096, 256, 0, stream>>>(encoder_out, sortind, enc_bf);
  mean_kernel<<<B_, 256, 0, stream>>>(encoder_out, sortind, mean_bf);

  transpose_bf16_kernel<<<2048, 256, 0, stream>>>(W_enc_att, Wenc_t, E_, A_);
  transpose_bf16_kernel<<<2048, 256, 0, stream>>>(W_dec_att, Wdec_t, D_, A_);
  transpose_bf16_kernel<<<4096, 256, 0, stream>>>(W_fbeta, Wfbeta_t, D_, E_);
  transpose_bf16_kernel<<<4096, 256, 0, stream>>>(W_fc, Wfc_t, D_, V_);
  transpose_bf16_kernel<<<4096, 256, 0, stream>>>(W_init_h, Winith_t, E_, D_);
  transpose_bf16_kernel<<<4096, 256, 0, stream>>>(W_init_c, Winitc_t, E_, D_);
  build_wcomb_kernel<<<4096, 256, 0, stream>>>(W_ih, W_hh, b_ih, b_hh, Wcomb, bcomb);

  // h0 / c0 = mean_enc @ W_init + b   (B x D, K=E)
  launch_gemm(stream, mean_bf, E_, Winith_t, E_, h_f, D_, b_init_h, nullptr, 0,
              B_, D_, E_, 0, h_bf, D_);
  launch_gemm(stream, mean_bf, E_, Winitc_t, E_, c_f, D_, b_init_c, nullptr, 0,
              B_, D_, E_, 0, nullptr, 0);

  // att1 = enc @ W_enc_att + b_enc_att   ((B*P) x A, K=E)  -- the big GEMM
  launch_gemm(stream, enc_bf, E_, Wenc_t, E_, att1, A_, b_enc_att, nullptr, 0,
              B_ * P_, A_, E_, 0, nullptr, 0);

  // ---- sequential decode ----
  for (int t = 0; t < TM1; ++t) {
    // att2 = h @ W_dec_att + b_dec_att
    launch_gemm(stream, h_bf, D_, Wdec_t, D_, att2, A_, b_dec_att, nullptr, 0,
                B_, A_, D_, 0, nullptr, 0);
    // e, softmax, alphas(out), awe
    attention_step<<<B_, 256, 0, stream>>>(att1, att2, w_full, b_full, encoder_out,
                                           sortind, declen, t, awe, out_alphas);
    // gate = sigmoid(h @ W_fbeta + b_fbeta)
    launch_gemm(stream, h_bf, D_, Wfbeta_t, D_, gateb, E_, b_fbeta, nullptr, 0,
                B_, E_, D_, 1, nullptr, 0);
    // xh = [e_t | gate*awe | h]
    build_x_kernel<<<(B_ * KC + 255) / 256, 256, 0, stream>>>(embs_bf, gateb, awe, h_bf, xh, t);
    // gates = xh @ [W_ih | W_hh]^T + (b_ih + b_hh)
    launch_gemm(stream, xh, KC, Wcomb, KC, gatesb, 4 * D_, bcomb, nullptr, 0,
                B_, 4 * D_, KC, 0, nullptr, 0);
    // LSTM pointwise + mask, update h/c (+ bf16 h)
    lstm_step<<<(B_ * D_ + 255) / 256, 256, 0, stream>>>(gatesb, h_f, c_f, h_bf, declen, t);
    // preds = mask * (h @ W_fc + b_fc), written straight into d_out
    launch_gemm(stream, h_bf, D_, Wfc_t, D_, out_pred + (size_t)t * V_,
                (long long)TM1 * V_, b_fc, declen, t, B_, V_, D_, 0, nullptr, 0);
  }
}
